// ReduceContextAttentionP2_17669495456478
// MI455X (gfx1250) — matrix operand order repaired and checked by reading, not compiled, not run on hardware
//
#include <hip/hip_runtime.h>
#include <stdint.h>

// ---------------------------------------------------------------------------
// B=4, C=32, H=W=256, K=16, stride 8, PD=4 -> hs=ws=32, N=1024 patches,
// padded 264x264. Dominant op: per-batch bf16 GEMM
//   [M=1024 pq] x [K=1024 n] x [N=33*256 (c,ij)]   (channel 32 == mask plane)
// then overlap-add scatter + crop (+ /wm), plus argmax-gather for recon.
// ---------------------------------------------------------------------------

typedef __attribute__((ext_vector_type(16))) __bf16 v16bf;
typedef __attribute__((ext_vector_type(8)))  float  v8f;
typedef __attribute__((ext_vector_type(4)))  int    v4i_;

union Frag {
  uint4 q[2];
  v16bf v;
};

__device__ __forceinline__ uint16_t f2bf(float f) {
  union { float f; uint32_t u; } v; v.f = f;
  uint32_t u = v.u;
  return (uint16_t)((u + 0x7FFFu + ((u >> 16) & 1u)) >> 16);  // RNE
}

__device__ __forceinline__ int clampi(int v, int lo, int hi) {
  return v < lo ? lo : (v > hi ? hi : v);
}

// ---- CDNA5 async global->LDS copy (signature confirmed by hipcc) -----------
#if defined(__has_builtin)
#  if __has_builtin(__builtin_amdgcn_global_load_async_to_lds_b128)
#    define USE_ASYNC_LDS 1
#  endif
#endif

#ifdef USE_ASYNC_LDS
typedef __attribute__((address_space(1))) v4i_* as1_v4i;
typedef __attribute__((address_space(3))) v4i_* as3_v4i;
#endif

__device__ __forceinline__ void cp_b128(uint16_t* ldst, const uint16_t* gsrc) {
#ifdef USE_ASYNC_LDS
  __builtin_amdgcn_global_load_async_to_lds_b128((as1_v4i)gsrc, (as3_v4i)ldst, 0, 0);
#else
  *(uint4*)ldst = *(const uint4*)gsrc;
#endif
}

__device__ __forceinline__ void async_wait_all() {
#ifdef USE_ASYNC_LDS
#  if __has_builtin(__builtin_amdgcn_s_wait_asynccnt)
  __builtin_amdgcn_s_wait_asynccnt(0);
#  else
  asm volatile("s_wait_asynccnt 0" ::: "memory");
#  endif
#endif
}

// ---- unfold once: Bmat[b][col][n] bf16, col = c*256+ij (c==32 -> mask) ------
__global__ void k_prep_bmat(const float* __restrict__ bg, const float* __restrict__ mask,
                            uint16_t* __restrict__ Bmat) {
  size_t gid = (size_t)blockIdx.x * 256 + threadIdx.x;
  if (gid >= (size_t)4 * 8448 * 1024) return;
  int n = (int)(gid & 1023);
  size_t t = gid >> 10;
  int col = (int)(t % 8448);
  int bb  = (int)(t / 8448);
  int c = col >> 8, ij = col & 255;
  int ii = ij >> 4, jj = ij & 15;
  int pn = n >> 5,  qn = n & 31;
  int ys = clampi(8 * pn + ii - 4, 0, 255);
  int xs = clampi(8 * qn + jj - 4, 0, 255);
  float m = mask[(bb << 16) + (ys << 8) + xs];
  float v = (c < 32) ? bg[(((size_t)(bb * 32 + c)) << 16) + (ys << 8) + xs] * (1.0f - m) : m;
  Bmat[gid] = f2bf(v);
}

// ---- cos [b][n][pq] f32  ->  cosT [b][pq][n] bf16 (LDS tile transpose) ------
__global__ void k_transpose_cos(const float* __restrict__ cs, uint16_t* __restrict__ cosT) {
  __shared__ uint16_t tile[32][33];
  int bb  = blockIdx.z;
  int pq0 = blockIdx.x * 32;
  int n0  = blockIdx.y * 32;
  int tx = threadIdx.x;   // 0..31
  int ty = threadIdx.y;   // 0..7
  for (int r = 0; r < 4; ++r) {
    int n = n0 + ty + r * 8;
    tile[ty + r * 8][tx] = f2bf(cs[(((size_t)(bb * 1024 + n)) << 10) + pq0 + tx]);
  }
  __syncthreads();
  for (int r = 0; r < 4; ++r) {
    int pq = pq0 + ty + r * 8;
    cosT[(((size_t)(bb * 1024 + pq)) << 10) + n0 + tx] = tile[tx][ty + r * 8];
  }
}

__global__ void k_zero(float* __restrict__ p, int n) {
  int gid = blockIdx.x * 256 + threadIdx.x;
  if (gid < n) p[gid] = 0.0f;
}

// ---- argmax over n for each (b, pq); coalesced over pq ----------------------
__global__ void k_argmax(const float* __restrict__ cs, int* __restrict__ idx) {
  int bb = blockIdx.y;
  int pq = blockIdx.x * 256 + threadIdx.x;   // grid.x = 4
  float best = -3.4e38f;
  int bi = 0;
  const float* base = cs + (((size_t)bb) << 20) + pq;
  for (int n = 0; n < 1024; ++n) {
    float v = base[((size_t)n) << 10];
    if (v > best) { best = v; bi = n; }
  }
  idx[bb * 1024 + pq] = bi;
}

// ---- main bf16 WMMA GEMM, double-buffered LDS, overlap-add epilogue ---------
#define MT  128
#define NT  256
#define LDA 40
#define LDB 40

__global__ void __launch_bounds__(256)
k_gemm(const uint16_t* __restrict__ cosT, const uint16_t* __restrict__ Bmat,
       float* __restrict__ acc_out, float* __restrict__ acc_m) {
  __shared__ __align__(16) uint16_t As[2][MT * LDA];   // [m][k]
  __shared__ __align__(16) uint16_t Bs[2][NT * LDB];   // [col][k]

  const int tid  = threadIdx.x;
  const int lane = tid & 31;
  const int wid  = tid >> 5;
  const int wm   = wid >> 2;   // 0..1 (64 rows each)
  const int wn   = wid & 3;    // 0..3 (64 cols each)

  const int c   = blockIdx.x;        // 0..32 (channel; 32 == mask plane)
  const int pq0 = blockIdx.y * MT;
  const int bb  = blockIdx.z;

  // staging source rows (K-contiguous bf16 in both operands)
  const int am = tid >> 1, aseg = tid & 1;
  const uint16_t* aRow = cosT + (((size_t)(bb * 1024 + pq0 + am)) << 10) + aseg * 16;
  const uint16_t* bRow = Bmat + (((size_t)(bb * 8448 + c * 256 + tid)) << 10);

  v8f acc[4][4] = {};

  auto stage = [&](int buf, int kt) {
    const int k0 = kt * 32;
    cp_b128(&As[buf][am * LDA + aseg * 16],     aRow + k0);
    cp_b128(&As[buf][am * LDA + aseg * 16 + 8], aRow + k0 + 8);
#pragma unroll
    for (int s = 0; s < 4; ++s)
      cp_b128(&Bs[buf][tid * LDB + s * 8], bRow + k0 + s * 8);
    __builtin_prefetch((const void*)(aRow + k0 + 64), 0, 2);
    __builtin_prefetch((const void*)(bRow + k0 + 64), 0, 2);
  };

  // fa[4] preloaded (32 VGPRs); fb loaded one 16x16 tile at a time (8 VGPRs)
  auto compute = [&](int buf) {
    Frag fa[4];
#pragma unroll
    for (int mm = 0; mm < 4; ++mm) {
      const uint16_t* rp = &As[buf][(wm * 64 + mm * 16 + (lane & 15)) * LDA];
      int ko = (lane < 16) ? 0 : 8;
      fa[mm].q[0] = *(const uint4*)(rp + ko);
      fa[mm].q[1] = *(const uint4*)(rp + ko + 16);
    }
#pragma unroll
    for (int nn = 0; nn < 4; ++nn) {
      Frag fb;
      const uint16_t* cp = &Bs[buf][(wn * 64 + nn * 16 + (lane & 15)) * LDB];
      int ko = (lane < 16) ? 0 : 8;
      fb.q[0] = *(const uint4*)(cp + ko);
      fb.q[1] = *(const uint4*)(cp + ko + 16);
#pragma unroll
      for (int mm = 0; mm < 4; ++mm)
        acc[mm][nn] = __builtin_amdgcn_wmma_f32_16x16x32_bf16(
            false, fa[mm].v, false, fb.v, (short)0, acc[mm][nn], false, false);
    }
  };

  stage(0, 0);
  async_wait_all();
  __syncthreads();
#pragma unroll 1
  for (int kt = 0; kt < 32; kt += 2) {
    // buffer index is a compile-time constant in each half -> no dynamic
    // LDS-base selects, minimal address pressure
    if (kt + 1 < 32) stage(1, kt + 1);
    compute(0);
    async_wait_all();
    __syncthreads();
    if (kt + 2 < 32) stage(0, kt + 2);
    compute(1);
    async_wait_all();
    __syncthreads();
  }

  // overlap-add scatter into padded 264x264 accumulators
#pragma unroll
  for (int mm = 0; mm < 4; ++mm) {
#pragma unroll
    for (int nn = 0; nn < 4; ++nn) {
#pragma unroll
      for (int r = 0; r < 8; ++r) {
        float val = acc[mm][nn][r];
        int M  = wm * 64 + mm * 16 + r + ((lane < 16) ? 0 : 8);
        int nl = wn * 64 + nn * 16 + (lane & 15);
        int pq = pq0 + M;
        int p = pq >> 5, q = pq & 31;
        int ii = nl >> 4, jj = nl & 15;
        int Y = 8 * p + ii, X = 8 * q + jj;
        if (c < 32) {
          atomicAdd(&acc_out[(((size_t)(bb * 32 + c)) * 264 + Y) * 264 + X], val);
        } else {
          atomicAdd(&acc_m[(((size_t)bb) * 264 + Y) * 264 + X], val);
        }
      }
    }
  }
}

// ---- recon: hardmax -> gather aux patch at argmax n, overlap-add ------------
__global__ void k_recon(const float* __restrict__ aux, const int* __restrict__ idx,
                        float* __restrict__ acc_r) {
  int bb = blockIdx.y;
  int pq = blockIdx.x;
  int n  = idx[bb * 1024 + pq];
  int pn = n >> 5, qn = n & 31;
  int pt = pq >> 5, qt = pq & 31;
  for (int e = threadIdx.x; e < 3 * 256; e += 256) {
    int c3 = e >> 8;
    int ij = e & 255;
    int ii = ij >> 4, jj = ij & 15;
    int ys = clampi(8 * pn + ii - 4, 0, 255);
    int xs = clampi(8 * qn + jj - 4, 0, 255);
    float v = aux[(((size_t)(bb * 3 + c3)) << 16) + (ys << 8) + xs];
    int Y = 8 * pt + ii, X = 8 * qt + jj;
    atomicAdd(&acc_r[(((size_t)(bb * 3 + c3)) * 264 + Y) * 264 + X], v);
  }
}

// ---- crop + divide by analytic overlap count wm -----------------------------
__device__ __forceinline__ float wcnt(int Y) {
  return (float)((Y < 256 ? 1 : 0) + (Y >= 8 ? 1 : 0));
}

__global__ void k_finalize(const float* __restrict__ acc_out, const float* __restrict__ acc_m,
                           const float* __restrict__ acc_r, float* __restrict__ out) {
  const size_t S0 = 8388608, S1 = 262144, S2 = 786432;
  size_t gid = (size_t)blockIdx.x * 256 + threadIdx.x;
  if (gid < S0) {
    size_t t = gid;
    int x = (int)(t & 255); t >>= 8;
    int y = (int)(t & 255); t >>= 8;
    int c = (int)(t & 31);  int bb = (int)(t >> 5);
    out[gid] = acc_out[(((size_t)(bb * 32 + c)) * 264 + (y + 4)) * 264 + (x + 4)];
  } else if (gid < S0 + S1) {
    size_t t = gid - S0;
    int x = (int)(t & 255); t >>= 8;
    int y = (int)(t & 255); int bb = (int)(t >> 8);
    float w = wcnt(y + 4) * wcnt(x + 4);
    out[gid] = acc_m[(((size_t)bb) * 264 + (y + 4)) * 264 + (x + 4)] / w;
  } else if (gid < S0 + S1 + S2) {
    size_t t = gid - S0 - S1;
    int x = (int)(t & 255); t >>= 8;
    int y = (int)(t & 255); t >>= 8;
    int c3 = (int)(t % 3); int bb = (int)(t / 3);
    float w = wcnt(y + 4) * wcnt(x + 4);
    out[gid] = acc_r[(((size_t)(bb * 3 + c3)) * 264 + (y + 4)) * 264 + (x + 4)] / w;
  }
}

// ---------------------------------------------------------------------------
extern "C" void kernel_launch(void* const* d_in, const int* in_sizes, int n_in,
                              void* d_out, int out_size, void* d_ws, size_t ws_size,
                              hipStream_t stream) {
  const float* cs   = (const float*)d_in[0];  // cos_similar (4,1024,32,32)
  const float* bg   = (const float*)d_in[1];  // b           (4,32,256,256)
  const float* mask = (const float*)d_in[2];  // mask        (4,1,256,256)
  const float* aux  = (const float*)d_in[3];  // aux         (4,3,256,256)
  float* out = (float*)d_out;

  char* ws = (char*)d_ws;
  uint16_t* Bmat   = (uint16_t*)(ws);                   // 69,206,016 B
  uint16_t* cosT   = (uint16_t*)(ws + 69206016);        //  8,388,608 B
  int*      idx    = (int*)     (ws + 77594624);        //     16,384 B
  float*    acc_out= (float*)   (ws + 77611008);        // 35,684,352 B
  float*    acc_m  = (float*)   (ws + 113295360);       //  1,115,136 B
  float*    acc_r  = (float*)   (ws + 114410496);       //  3,345,408 B (end ~118 MB)

  k_prep_bmat<<<(int)(((size_t)4 * 8448 * 1024 + 255) / 256), 256, 0, stream>>>(bg, mask, Bmat);
  k_transpose_cos<<<dim3(32, 32, 4), dim3(32, 8), 0, stream>>>(cs, cosT);
  const int nz = 4 * 36 * 69696;  // acc_out + acc_m + acc_r (contiguous)
  k_zero<<<(nz + 255) / 256, 256, 0, stream>>>(acc_out, nz);
  k_argmax<<<dim3(4, 4), 256, 0, stream>>>(cs, idx);
  k_gemm<<<dim3(33, 8, 4), 256, 0, stream>>>(cosT, Bmat, acc_out, acc_m);
  k_recon<<<dim3(1024, 4), 256, 0, stream>>>(aux, idx, acc_r);
  k_finalize<<<(9437184 + 255) / 256, 256, 0, stream>>>(acc_out, acc_m, acc_r, out);
}
